// BiLSTM_CRF_38062000177831
// MI455X (gfx1250) — matrix operand order repaired and checked by reading, not compiled, not run on hardware
//
#include <hip/hip_runtime.h>
#include <hip/hip_bf16.h>
#include <math.h>

// ---------------------------------------------------------------------------
// BiLSTM-CRF for MI455X (gfx1250).
//   T=4096, IN=2048, H=1024 (per dir), 4H=4096, K=50 tags (START=48, STOP=49)
// ---------------------------------------------------------------------------

typedef __attribute__((ext_vector_type(8)))  __bf16 v8bf;
typedef __attribute__((ext_vector_type(16))) __bf16 v16bf;
typedef __attribute__((ext_vector_type(8)))  float  v8f;

#define T_LEN   4096
#define IN_DIM  2048
#define HDIM    1024
#define G4      4096        // 4*H gate rows
#define KTAG    50
#define KPAD    64
#define START_T 48
#define STOP_T  49
#define NEGV    (-10000.0f)
#define NBLK_PER_DIR 16
#define NBLK_LSTM    32     // both directions

__device__ __forceinline__ float sigmoidf_(float x) {
    return 1.0f / (1.0f + expf(-x));
}

// ------------------------- grid-wide barrier -------------------------------
__device__ __forceinline__ void grid_barrier(unsigned int* cnt, unsigned int* gen,
                                             unsigned int nblk) {
    __syncthreads();
    if (threadIdx.x == 0) {
        __threadfence();
        unsigned int g = __hip_atomic_load(gen, __ATOMIC_RELAXED, __HIP_MEMORY_SCOPE_AGENT);
        unsigned int arrived =
            __hip_atomic_fetch_add(cnt, 1u, __ATOMIC_ACQ_REL, __HIP_MEMORY_SCOPE_AGENT);
        if (arrived == nblk - 1u) {
            __hip_atomic_store(cnt, 0u, __ATOMIC_RELAXED, __HIP_MEMORY_SCOPE_AGENT);
            __hip_atomic_fetch_add(gen, 1u, __ATOMIC_RELEASE, __HIP_MEMORY_SCOPE_AGENT);
        } else {
            while (__hip_atomic_load(gen, __ATOMIC_ACQUIRE, __HIP_MEMORY_SCOPE_AGENT) == g)
                __builtin_amdgcn_s_sleep(1);
        }
        __threadfence();
    }
    __syncthreads();
}

// ------------------------- prep kernels ------------------------------------
__global__ void f2bf_kernel(const float* __restrict__ src, __bf16* __restrict__ dst, int n) {
    int i = blockIdx.x * blockDim.x + threadIdx.x;
    if (i < n) dst[i] = (__bf16)src[i];
}

// wT[k*4096 + r] = w_hh[r*1024 + k]  (k-major so GEMV lanes are coalesced in r)
__global__ void transpose_whh_kernel(const float* __restrict__ w, float* __restrict__ wT) {
    int i = blockIdx.x * blockDim.x + threadIdx.x;   // i < 4096*1024
    int r = i & (G4 - 1);
    int k = i >> 12;
    wT[i] = w[(size_t)r * HDIM + k];
}

// wOutT[k*64 + n] = w_out[n*2048 + k] (padded to 64 cols)
__global__ void prep_wout_kernel(const float* __restrict__ w_out, float* __restrict__ wOutT) {
    int i = blockIdx.x * blockDim.x + threadIdx.x;   // i < 2048*64
    int n = i & (KPAD - 1);
    int k = i >> 6;
    wOutT[i] = (n < KTAG) ? w_out[(size_t)n * (2 * HDIM) + k] : 0.0f;
}

__global__ void init_bar_kernel(unsigned int* bar) {
    bar[0] = 0u;
    bar[1] = 0u;
}

// ------------------------- WMMA input-projection GEMM ----------------------
// xg[dir][t, r] = sum_k sent'[t,k] * w_ih[dir][r,k] + b[dir][r]
// sent' = sentence for dir 0, reversed sentence for dir 1.
// Per wave: one 16(M) x 64(N) tile, K-loop in chunks of 32 (bf16 WMMA).
__device__ __forceinline__ v16bf load16bf(const __bf16* p) {
    v8bf lo = *(const v8bf*)p;
    v8bf hi = *(const v8bf*)(p + 8);
    return __builtin_shufflevector(lo, hi, 0,1,2,3,4,5,6,7,8,9,10,11,12,13,14,15);
}

__global__ __launch_bounds__(256) void xg_gemm_kernel(
    const __bf16* __restrict__ sent_bf,
    const __bf16* __restrict__ wih_f_bf,
    const __bf16* __restrict__ wih_b_bf,
    const float*  __restrict__ b_f,
    const float*  __restrict__ b_b,
    float* __restrict__ xg_f,
    float* __restrict__ xg_b)
{
    const int dir = blockIdx.z;
    const __bf16* __restrict__ wih  = dir ? wih_b_bf : wih_f_bf;
    const float*  __restrict__ bias = dir ? b_b : b_f;
    float* __restrict__ xg          = dir ? xg_b : xg_f;

    const int wave = threadIdx.x >> 5;
    const int lane = threadIdx.x & 31;
    const int half = lane >> 4;      // which lane-half (K split)
    const int l15  = lane & 15;

    const int m0 = blockIdx.y * 16;
    const int n0 = (blockIdx.x * 8 + wave) * 64;

    // A fragment: lane = row M; lanes 0-15 hold K kc..kc+15,
    //             lanes 16-31 hold K kc+16..kc+31.
    const int arow = dir ? (T_LEN - 1 - (m0 + l15)) : (m0 + l15);
    const __bf16* __restrict__ aptr = sent_bf + (size_t)arow * IN_DIM + (half ? 16 : 0);

    // B fragment: lane = column N (w_ih row); lanes 0-15 hold K kc..kc+15,
    //             lanes 16-31 hold K kc+16..kc+31 (contiguous in w row).
    const __bf16* bp[4];
    v8f acc[4];
#pragma unroll
    for (int s = 0; s < 4; ++s) {
        bp[s] = wih + (size_t)(n0 + s * 16 + l15) * IN_DIM + (half ? 16 : 0);
        float bv = bias[n0 + s * 16 + l15];
        acc[s] = (v8f){bv, bv, bv, bv, bv, bv, bv, bv};
    }

    for (int kc = 0; kc < IN_DIM; kc += 32) {
        v8bf alo = *(const v8bf*)(aptr + kc);
        v8bf ahi = *(const v8bf*)(aptr + kc + 8);
        v16bf a = __builtin_shufflevector(alo, ahi,
                                          0,1,2,3,4,5,6,7,8,9,10,11,12,13,14,15);
#pragma unroll
        for (int s = 0; s < 4; ++s) {
            v16bf b = load16bf(bp[s] + kc);
            acc[s] = __builtin_amdgcn_wmma_f32_16x16x32_bf16(
                false, a, false, b, (short)0, acc[s], false, false);
        }
    }

    // C/D layout: VGPR v -> M = v (lanes 0-15) / v+8 (lanes 16-31), N = lane&15
    const int mbase = m0 + (half ? 8 : 0);
#pragma unroll
    for (int s = 0; s < 4; ++s) {
        const int col = n0 + s * 16 + l15;
#pragma unroll
        for (int vi = 0; vi < 8; ++vi)
            xg[(size_t)(mbase + vi) * G4 + col] = acc[s][vi];
    }
}

// ------------------------- persistent bidirectional LSTM -------------------
// 32 workgroups (16 per direction), 256 threads each; each thread owns one
// gate row r = gate*1024 + unit; fp32 GEMV vs k-major wT streamed from L2;
// h ping-pong in global, staged to LDS; one grid barrier per step.
__global__ __launch_bounds__(256) void lstm_kernel(
    const float* __restrict__ xg_f, const float* __restrict__ xg_b,
    const float* __restrict__ wT_f, const float* __restrict__ wT_b,
    const float* __restrict__ h0,   const float* __restrict__ c0,
    float* __restrict__ hbuf,       // [dir][2][1024]
    float* __restrict__ hs_f, float* __restrict__ hs_b,
    unsigned int* __restrict__ bar)
{
    const int dir = blockIdx.x >> 4;
    const int wg  = blockIdx.x & 15;
    const float* __restrict__ xg = dir ? xg_b : xg_f;
    const float* __restrict__ wT = dir ? wT_b : wT_f;
    float* __restrict__ hs = dir ? hs_b : hs_f;
    float* hb0 = hbuf + dir * 2 * HDIM;
    float* hb1 = hb0 + HDIM;

    const int tid  = threadIdx.x;
    const int gate = tid >> 6;
    const int jloc = tid & 63;
    const int r    = gate * HDIM + wg * 64 + jloc;

    __shared__ float h_lds[HDIM];
    __shared__ float gbuf[256];

    float c = 0.0f;
    if (tid < 64) {
        int j = wg * 64 + tid;
        c = c0[dir * HDIM + j];
        hb0[j] = h0[dir * HDIM + j];
    }
    grid_barrier(bar, bar + 1, NBLK_LSTM);

    const float* __restrict__ wcol = wT + r;
    for (int t = 0; t < T_LEN; ++t) {
        const float* hin  = (t & 1) ? hb1 : hb0;
        float*       hout = (t & 1) ? hb0 : hb1;
        for (int i = tid; i < HDIM; i += 256) h_lds[i] = hin[i];
        __syncthreads();

        float acc = xg[(size_t)t * G4 + r];
        if (t + 1 < T_LEN)
            __builtin_prefetch(&xg[(size_t)(t + 1) * G4 + r], 0, 1);
#pragma unroll 8
        for (int k = 0; k < HDIM; ++k)
            acc = fmaf(h_lds[k], wcol[k * G4], acc);
        gbuf[tid] = acc;
        __syncthreads();

        if (tid < 64) {
            int j = wg * 64 + tid;
            float ig = sigmoidf_(gbuf[tid]);
            float fg = sigmoidf_(gbuf[64 + tid]);
            float gg = tanhf(gbuf[128 + tid]);
            float og = sigmoidf_(gbuf[192 + tid]);
            c = fg * c + ig * gg;
            float h = og * tanhf(c);
            hout[j] = h;
            hs[(size_t)t * HDIM + j] = h;
        }
        grid_barrier(bar, bar + 1, NBLK_LSTM);
    }
}

// ------------------------- output projection (feats) -----------------------
// feats[t, n] = [hs_f[t], hs_b[T-1-t]] . w_out[n] + b_out[n], fp32, K padded 64
__global__ __launch_bounds__(256) void feats_kernel(
    const float* __restrict__ hs_f, const float* __restrict__ hs_b,
    const float* __restrict__ wOutT, const float* __restrict__ b_out,
    float* __restrict__ feats)
{
    const int t = blockIdx.x;
    __shared__ float x[2 * HDIM];
    __shared__ float part[4][KPAD];
    const int tid = threadIdx.x;

    for (int i = tid; i < HDIM; i += 256) x[i] = hs_f[(size_t)t * HDIM + i];
    for (int i = tid; i < HDIM; i += 256) x[HDIM + i] = hs_b[(size_t)(T_LEN - 1 - t) * HDIM + i];
    __syncthreads();

    const int n  = tid & (KPAD - 1);
    const int ks = tid >> 6;
    float acc = 0.0f;
    const float* __restrict__ wp = wOutT + n;
    const int k0 = ks * 512;
#pragma unroll 4
    for (int k = k0; k < k0 + 512; ++k)
        acc = fmaf(x[k], wp[k * KPAD], acc);
    part[ks][n] = acc;
    __syncthreads();

    if (tid < KPAD) {
        float s = part[0][tid] + part[1][tid] + part[2][tid] + part[3][tid];
        feats[(size_t)t * KPAD + tid] = (tid < KTAG) ? (s + b_out[tid]) : -1.0e30f;
    }
}

// ------------------------- Viterbi (single WG) -----------------------------
__global__ __launch_bounds__(64) void viterbi_kernel(
    const float* __restrict__ feats, const float* __restrict__ trans,
    unsigned char* __restrict__ bptr, float* __restrict__ out)
{
    __shared__ float tr[KTAG * KTAG];
    __shared__ float fv[2][KPAD];
    const int tid = threadIdx.x;

    for (int i = tid; i < KTAG * KTAG; i += 64) tr[i] = trans[i];
    fv[0][tid] = (tid == START_T) ? 0.0f : NEGV;
    fv[1][tid] = NEGV;
    __syncthreads();

    int cur = 0;
    for (int t = 0; t < T_LEN; ++t) {
        float nf = NEGV;
        int   arg = 0;
        if (tid < KTAG) {
            float best = -3.4e38f;
            const float* trow = &tr[tid * KTAG];
#pragma unroll 5
            for (int p = 0; p < KTAG; ++p) {
                float s = fv[cur][p] + trow[p];
                if (s > best) { best = s; arg = p; }
            }
            nf = best + feats[(size_t)t * KPAD + tid];
            bptr[(size_t)t * KTAG + tid] = (unsigned char)arg;
        }
        if (tid < KTAG) fv[cur ^ 1][tid] = nf;
        cur ^= 1;
        __syncthreads();
    }

    if (tid == 0) {
        float best = -3.4e38f;
        int bl = 0;
        for (int p = 0; p < KTAG; ++p) {
            float s = fv[cur][p] + tr[STOP_T * KTAG + p];
            if (s > best) { best = s; bl = p; }
        }
        out[0] = best;                       // path_score
        out[1 + (T_LEN - 1)] = (float)bl;    // path[T-1]
        int tag = bl;
        for (int t = T_LEN - 1; t >= 1; --t) {
            tag = (int)bptr[(size_t)t * KTAG + tag];
            out[1 + t - 1] = (float)tag;
        }
    }
}

// ---------------------------------------------------------------------------
extern "C" void kernel_launch(void* const* d_in, const int* in_sizes, int n_in,
                              void* d_out, int out_size, void* d_ws, size_t ws_size,
                              hipStream_t stream)
{
    (void)in_sizes; (void)n_in; (void)out_size; (void)ws_size;

    const float* sentence    = (const float*)d_in[0];
    const float* w_ih_f      = (const float*)d_in[1];
    const float* w_hh_f      = (const float*)d_in[2];
    const float* b_f         = (const float*)d_in[3];
    const float* w_ih_b      = (const float*)d_in[4];
    const float* w_hh_b      = (const float*)d_in[5];
    const float* b_b         = (const float*)d_in[6];
    const float* w_out       = (const float*)d_in[7];
    const float* b_out       = (const float*)d_in[8];
    const float* transitions = (const float*)d_in[9];
    const float* h0          = (const float*)d_in[10];
    const float* c0          = (const float*)d_in[11];
    float* out = (float*)d_out;

    char* ws = (char*)d_ws;
    size_t off = 0;
    auto carve = [&](size_t bytes) -> void* {
        void* p = (void*)(ws + off);
        off += (bytes + 255) & ~(size_t)255;
        return p;
    };

    float*  xg_f    = (float*) carve((size_t)T_LEN * G4 * 4);       // 64 MB
    float*  xg_b    = (float*) carve((size_t)T_LEN * G4 * 4);       // 64 MB
    __bf16* sent_bf = (__bf16*)carve((size_t)T_LEN * IN_DIM * 2);   // 16 MB
    __bf16* wihf_bf = (__bf16*)carve((size_t)G4 * IN_DIM * 2);      // 16 MB
    __bf16* wihb_bf = (__bf16*)carve((size_t)G4 * IN_DIM * 2);      // 16 MB
    float*  wT_f    = (float*) carve((size_t)HDIM * G4 * 4);        // 16 MB
    float*  wT_b    = (float*) carve((size_t)HDIM * G4 * 4);        // 16 MB
    float*  hs_f    = (float*) carve((size_t)T_LEN * HDIM * 4);     // 16 MB
    float*  hs_b    = (float*) carve((size_t)T_LEN * HDIM * 4);     // 16 MB
    float*  wOutT   = (float*) carve((size_t)(2 * HDIM) * KPAD * 4);
    float*  feats   = (float*) carve((size_t)T_LEN * KPAD * 4);
    unsigned char* bptr = (unsigned char*)carve((size_t)T_LEN * KTAG);
    float*  hbuf    = (float*) carve((size_t)2 * 2 * HDIM * 4);
    unsigned int* bar = (unsigned int*)carve(256);

    const int nSent = T_LEN * IN_DIM;   // 8,388,608
    const int nWih  = G4 * IN_DIM;      // 8,388,608
    const int nWhh  = G4 * HDIM;        // 4,194,304

    // prep: bf16 conversions, weight transposes, barrier init
    f2bf_kernel<<<(nSent + 255) / 256, 256, 0, stream>>>(sentence, sent_bf, nSent);
    f2bf_kernel<<<(nWih + 255) / 256, 256, 0, stream>>>(w_ih_f, wihf_bf, nWih);
    f2bf_kernel<<<(nWih + 255) / 256, 256, 0, stream>>>(w_ih_b, wihb_bf, nWih);
    transpose_whh_kernel<<<nWhh / 256, 256, 0, stream>>>(w_hh_f, wT_f);
    transpose_whh_kernel<<<nWhh / 256, 256, 0, stream>>>(w_hh_b, wT_b);
    prep_wout_kernel<<<(2 * HDIM * KPAD) / 256, 256, 0, stream>>>(w_out, wOutT);
    init_bar_kernel<<<1, 1, 0, stream>>>(bar);

    // big bf16 WMMA GEMM: xg for both directions
    {
        dim3 grid(G4 / (8 * 64), T_LEN / 16, 2);   // (8, 256, 2)
        xg_gemm_kernel<<<grid, 256, 0, stream>>>(sent_bf, wihf_bf, wihb_bf,
                                                 b_f, b_b, xg_f, xg_b);
    }

    // persistent bidirectional LSTM recurrence (fp32)
    lstm_kernel<<<NBLK_LSTM, 256, 0, stream>>>(xg_f, xg_b, wT_f, wT_b,
                                               h0, c0, hbuf, hs_f, hs_b, bar);

    // output projection
    feats_kernel<<<T_LEN, 256, 0, stream>>>(hs_f, hs_b, wOutT, b_out, feats);

    // Viterbi decode + backtrace
    viterbi_kernel<<<1, 64, 0, stream>>>(feats, transitions, bptr, out);
}
